// LearnableMatchingCost_4930622455850
// MI455X (gfx1250) — matrix operand order repaired and checked by reading, not compiled, not run on hardware
//
#include <hip/hip_runtime.h>
#include <hip/hip_bf16.h>

typedef __attribute__((ext_vector_type(16))) __bf16 v16bf;
typedef __attribute__((ext_vector_type(8)))  float  v8f;

__device__ __forceinline__ unsigned short f32_to_bf16_rne(float f) {
    unsigned int u = __float_as_uint(f);
    unsigned int r = u + 0x7FFFu + ((u >> 16) & 1u);
    return (unsigned short)(r >> 16);
}

union BF16x16 { uint4 q[2]; v16bf v; };

// ---------------------------------------------------------------------------
// Cost volume -> NHWC bf16 activations: act[((n*64+h)*64+w)*64 + c]
// n = b*49 + u*7 + v.  cost[:32]=x*valid, cost[32:]=y_shift*valid,
// valid = (sum_c y_shift != 0).
// ---------------------------------------------------------------------------
__global__ __launch_bounds__(256) void build_cost(
    const float* __restrict__ x, const float* __restrict__ y,
    unsigned short* __restrict__ act)
{
    int idx = blockIdx.x * 256 + threadIdx.x;
    if (idx >= 196 * 4096) return;
    int n   = idx >> 12;
    int pix = idx & 4095;
    int h = pix >> 6, w = pix & 63;
    int b = n / 49,  s = n % 49;
    int i = s / 7,   j = s % 7;
    int ind  = i - 3;   // width shift
    int indd = j - 3;   // height shift
    int hs = h + indd, ws = w + ind;
    bool inb = (hs >= 0) && (hs < 64) && (ws >= 0) && (ws < 64);

    float sum = 0.f;
    if (inb) {
        for (int c = 0; c < 32; ++c)
            sum += y[((b * 32 + c) * 64 + hs) * 64 + ws];
    }
    float mask = (inb && (sum != 0.f)) ? 1.f : 0.f;

    unsigned short pk[64];
#pragma unroll
    for (int c = 0; c < 32; ++c) {
        float xv = x[((b * 32 + c) * 64 + h) * 64 + w] * mask;
        float yv = inb ? (y[((b * 32 + c) * 64 + hs) * 64 + ws] * mask) : 0.f;
        pk[c]      = f32_to_bf16_rne(xv);
        pk[c + 32] = f32_to_bf16_rne(yv);
    }
    uint4* dst = (uint4*)&act[(size_t)idx * 64];
#pragma unroll
    for (int q = 0; q < 8; ++q) {
        uint4 v;
        __builtin_memcpy(&v, &pk[q * 8], 16);
        dst[q] = v;
    }
}

// ---------------------------------------------------------------------------
// Weight reorder + f32->bf16:  dst[co][(r*3+c)*CIN + ci] = W[co][ci][r][c]
// Rows padded to COUTP (multiple of 16), extra rows zero.
// ---------------------------------------------------------------------------
__global__ __launch_bounds__(256) void cvt_weights(
    const float* __restrict__ src, unsigned short* __restrict__ dst,
    int COUT, int COUTP, int CIN)
{
    int K = CIN * 9;
    int i = blockIdx.x * 256 + threadIdx.x;
    if (i >= COUTP * K) return;
    int co  = i / K;
    int rem = i - co * K;
    int rc  = rem / CIN;          // r*3 + c
    int ci  = rem - rc * CIN;
    float v = 0.f;
    if (co < COUT) v = src[(co * CIN + ci) * 9 + rc];   // OIHW flat
    dst[i] = f32_to_bf16_rne(v);
}

// ---------------------------------------------------------------------------
// Implicit-GEMM 3x3 SAME conv + bias + ReLU, NHWC, v_wmma_f32_16x16x32_bf16.
// Workgroup = (image n, row h), 4 waves. LDS = [3 rows][66 cols][CIN] bf16,
// staged with global_load_async_to_lds_b128 (halo zero-filled via ds stores).
// Each wave iteration: one 16-co weight tile x TWO 16-pixel tiles ->
// shared A operand, two independent WMMA accumulator chains.
// ---------------------------------------------------------------------------
template <int CIN, int COUT, bool LAST>
__global__ __launch_bounds__(128) void conv3x3_wmma(
    const unsigned short* __restrict__ in,    // [N,64,64,CIN]  bf16 NHWC
    const unsigned short* __restrict__ wts,   // [COUTP, CIN*9] bf16 (reordered)
    const float* __restrict__ bias,           // [COUT]
    unsigned short* __restrict__ out_bf,      // [N,64,64,COUT] bf16 NHWC
    float* __restrict__ out_f32)              // [N,64,64] f32 (LAST only)
{
    constexpr int K     = CIN * 9;
    constexpr int LDS_W = 66;
    constexpr int C8    = CIN / 8;            // 16B chunks per pixel
    __shared__ alignas(16) unsigned short lds[3 * LDS_W * CIN];

    const int h   = blockIdx.x;
    const int n   = blockIdx.y;
    const int tid = threadIdx.x;

    // ---- stage rows h-1..h+1 (halo cols -1,64) as 16B chunks ----
    constexpr int NCH = 3 * LDS_W * C8;
    for (int idx = tid; idx < NCH; idx += 128) {
        int c8   = idx % C8;
        int rest = idx / C8;
        int col  = rest % LDS_W;
        int r    = rest / LDS_W;
        int hi   = h - 1 + r;
        int wi   = col - 1;
        unsigned lds_half = (unsigned)((r * LDS_W + col) * CIN + c8 * 8);
        if (hi >= 0 && hi < 64 && wi >= 0 && wi < 64) {
            unsigned long long g =
                (unsigned long long)(uintptr_t)(in +
                    ((size_t)((n * 64 + hi) * 64 + wi) * CIN + c8 * 8));
            unsigned loff = (unsigned)(uintptr_t)(&lds[lds_half]);
            asm volatile("global_load_async_to_lds_b128 %0, %1, off"
                         :: "v"(loff), "v"(g) : "memory");
        } else {
            uint4 z = {0u, 0u, 0u, 0u};
            *(uint4*)(&lds[lds_half]) = z;
        }
    }
    asm volatile("s_wait_asynccnt 0" ::: "memory");
    __syncthreads();

    const int  wave   = tid >> 5;
    const int  lane   = tid & 31;
    const int  row    = lane & 15;
    const bool hiHalf = lane >= 16;
    const int  khalf  = hiHalf ? 16 : 0;      // B K-offset for this half
    const int  ahalf  = hiHalf ? 1  : 0;      // A uint4 selector

    constexpr int MT     = (COUT + 15) / 16;
    constexpr int TILES2 = MT * 2;            // x 2 pixel-tile pairs

    for (int tile = wave; tile < TILES2; tile += 4) {
        const int mbase = (tile >> 1) * 16;
        const int pbase = (tile & 1) * 32;
        const int p0    = pbase + row;        // pixel tile 0
        const int p1    = pbase + 16 + row;   // pixel tile 1
        const unsigned short* wrow = wts + (size_t)(mbase + row) * K;

        v8f acc0 = {};
        v8f acc1 = {};
#pragma unroll 3
        for (int k0 = 0; k0 < K; k0 += 32) {
            const int rc  = k0 / CIN;         // tap index r*3+c (const in chunk)
            const int ci0 = k0 - rc * CIN;
            const int rr  = rc / 3;
            const int cc  = rc - rr * 3;
            const int rowbase = (rr * LDS_W + cc) * CIN + ci0 + khalf;
            // A: weight row (shared by both pixel tiles)
            const uint4* ap = (const uint4*)(wrow + k0);
            BF16x16 au;
            au.q[0] = ap[2 * ahalf];
            au.q[1] = ap[2 * ahalf + 1];
            // B0 / B1: 16 contiguous channels for each pixel tile
            const uint4* bp0 = (const uint4*)(&lds[rowbase + p0 * CIN]);
            const uint4* bp1 = (const uint4*)(&lds[rowbase + p1 * CIN]);
            BF16x16 bu0, bu1;
            bu0.q[0] = bp0[0]; bu0.q[1] = bp0[1];
            bu1.q[0] = bp1[0]; bu1.q[1] = bp1[1];
            acc0 = __builtin_amdgcn_wmma_f32_16x16x32_bf16(
                false, au.v, false, bu0.v, (short)0, acc0, false, false);
            acc1 = __builtin_amdgcn_wmma_f32_16x16x32_bf16(
                false, au.v, false, bu1.v, (short)0, acc1, false, false);
        }

        // ---- epilogue: C layout -> lane pixel p, co = mbase+8*hiHalf+i ----
        const int coff = mbase + (hiHalf ? 8 : 0);
        if (LAST) {
            if (!hiHalf) {
                float v0 = fmaxf(acc0[0] + bias[0], 0.f);
                float v1 = fmaxf(acc1[0] + bias[0], 0.f);
                out_f32[(size_t)(n * 64 + h) * 64 + p0] = v0;
                out_f32[(size_t)(n * 64 + h) * 64 + p1] = v1;
            }
        } else {
            unsigned short pk0[8], pk1[8];
#pragma unroll
            for (int i = 0; i < 8; ++i) {
                float b  = bias[coff + i];
                pk0[i] = f32_to_bf16_rne(fmaxf(acc0[i] + b, 0.f));
                pk1[i] = f32_to_bf16_rne(fmaxf(acc1[i] + b, 0.f));
            }
            uint4 st0, st1;
            __builtin_memcpy(&st0, pk0, 16);
            __builtin_memcpy(&st1, pk1, 16);
            *(uint4*)(&out_bf[(size_t)((n * 64 + h) * 64 + p0) * COUT + coff]) = st0;
            *(uint4*)(&out_bf[(size_t)((n * 64 + h) * 64 + p1) * COUT + coff]) = st1;
        }
    }
}

// ---------------------------------------------------------------------------
extern "C" void kernel_launch(void* const* d_in, const int* in_sizes, int n_in,
                              void* d_out, int out_size, void* d_ws, size_t ws_size,
                              hipStream_t stream) {
    (void)in_sizes; (void)n_in; (void)out_size; (void)ws_size;

    const float* x = (const float*)d_in[0];
    const float* y = (const float*)d_in[1];
    const float* W[6]; const float* B[6];
    for (int l = 0; l < 6; ++l) {
        W[l] = (const float*)d_in[2 + 2 * l];
        B[l] = (const float*)d_in[3 + 2 * l];
    }

    static const int CI[6]  = { 64, 96, 128, 128, 64, 32 };
    static const int CO[6]  = { 96, 128, 128, 64, 32, 1 };
    static const int COP[6] = { 96, 128, 128, 64, 32, 16 };   // padded rows

    const size_t ACT_ELEMS = (size_t)196 * 128 * 4096;        // max 128 ch
    char* ws = (char*)d_ws;
    size_t off = 0;
    auto align256 = [](size_t v) { return (v + 255) & ~(size_t)255; };

    unsigned short* actA = (unsigned short*)(ws + off); off += align256(ACT_ELEMS * 2);
    unsigned short* actB = (unsigned short*)(ws + off); off += align256(ACT_ELEMS * 2);

    unsigned short* wbf[6];
    for (int l = 0; l < 6; ++l) {
        wbf[l] = (unsigned short*)(ws + off);
        off += align256((size_t)COP[l] * CI[l] * 9 * 2);
    }

    for (int l = 0; l < 6; ++l) {
        int total = COP[l] * CI[l] * 9;
        cvt_weights<<<(total + 255) / 256, 256, 0, stream>>>(
            W[l], wbf[l], CO[l], COP[l], CI[l]);
    }

    {
        int total = 196 * 4096;
        build_cost<<<(total + 255) / 256, 256, 0, stream>>>(x, y, actA);
    }

    dim3 grid(64, 196, 1);
    dim3 blk(128, 1, 1);
    conv3x3_wmma< 64,  96, false><<<grid, blk, 0, stream>>>(actA, wbf[0], B[0], actB, nullptr);
    conv3x3_wmma< 96, 128, false><<<grid, blk, 0, stream>>>(actB, wbf[1], B[1], actA, nullptr);
    conv3x3_wmma<128, 128, false><<<grid, blk, 0, stream>>>(actA, wbf[2], B[2], actB, nullptr);
    conv3x3_wmma<128,  64, false><<<grid, blk, 0, stream>>>(actB, wbf[3], B[3], actA, nullptr);
    conv3x3_wmma< 64,  32, false><<<grid, blk, 0, stream>>>(actA, wbf[4], B[4], actB, nullptr);
    conv3x3_wmma< 32,   1, true ><<<grid, blk, 0, stream>>>(actB, wbf[5], B[5], nullptr, (float*)d_out);
}
